// MHA_76175539962578
// MI455X (gfx1250) — matrix operand order repaired and checked by reading, not compiled, not run on hardware
//
#include <hip/hip_runtime.h>
#include <hip/hip_bf16.h>

typedef __attribute__((ext_vector_type(16))) _Float16 v16h;
typedef __attribute__((ext_vector_type(8)))  _Float16 h8;
typedef __attribute__((ext_vector_type(8)))  float    v8f;

#define SFULL  2048
#define DMODEL 1024
#define NHEAD  16
#define DHEAD  64

// Build a 16-wide f16 A/B fragment from two contiguous 8-half (16B) chunks.
__device__ __forceinline__ v16h make_frag(const _Float16* p0, const _Float16* p1) {
  h8 lo = *(const h8*)p0;
  h8 hi = *(const h8*)p1;
  return __builtin_shufflevector(lo, hi, 0,1,2,3,4,5,6,7,8,9,10,11,12,13,14,15);
}

// CDNA5 async memory->LDS copy (16B per lane), tracked by ASYNCcnt.
// vdst = LDS byte address (low 32 bits of generic pointer), voff = global byte
// offset, saddr = 64-bit SGPR base.  GVS addressing mode (§08 async tensor).
__device__ __forceinline__ void async_ld_lds_b128(unsigned lds_addr,
                                                  const void* sbase,
                                                  unsigned gbyteoff) {
  asm volatile("global_load_async_to_lds_b128 %0, %1, %2"
               :
               : "v"(lds_addr), "v"(gbyteoff), "s"(sbase)
               : "memory");
}
__device__ __forceinline__ void wait_async_all() {
  asm volatile("s_wait_asynccnt 0x0" ::: "memory");
}

// ---------------------------------------------------------------------------
// GEMM: Out[b,h,s,d] (f16) = X[m,:] @ W[n,:]^T + bias[n]
// X: [4096, 1024] f32, W: [1024, 1024] f32 (row n holds K), out in BHSD f16.
// Block: 256 thr (8 waves), tile 128(M) x 64(N); wave = 32x32 (2x2 WMMA accs).
// f32 -> f16 conversion forces the VGPR staging path here (async copy cannot
// convert), so only prefetch is added on this path.
// ---------------------------------------------------------------------------
__global__ __launch_bounds__(256) void proj_qkv_wmma(
    const float* __restrict__ X, const float* __restrict__ W,
    const float* __restrict__ bias, _Float16* __restrict__ Out)
{
  __shared__ __align__(16) _Float16 As[128][32];
  __shared__ __align__(16) _Float16 Bs[64][32];

  const int t    = threadIdx.x;
  const int lane = t & 31;
  const int wave = t >> 5;
  const int wm = wave & 3, wn = wave >> 2;
  const int Mbase = blockIdx.x * 128;
  const int Nbase = blockIdx.y * 64;
  const int lo16 = lane & 15;
  const int hi   = (lane >> 4) & 1;

  v8f acc[2][2];
  #pragma unroll
  for (int i = 0; i < 2; i++)
    #pragma unroll
    for (int j = 0; j < 2; j++) { v8f z = {}; acc[i][j] = z; }

  for (int k0 = 0; k0 < DMODEL; k0 += 32) {
    // L2 prefetch of the next K-chunk tiles.
    if (k0 + 32 < DMODEL) {
      __builtin_prefetch(&X[(size_t)(Mbase + (t >> 1)) * DMODEL + k0 + 32], 0, 1);
      if (t < 64)
        __builtin_prefetch(&W[(size_t)(Nbase + t) * DMODEL + k0 + 32], 0, 1);
    }
    #pragma unroll
    for (int i = 0; i < 16; i++) {             // 128x32 A tile, f32 -> f16
      int idx = t + i * 256;
      int r = idx >> 5, c = idx & 31;
      As[r][c] = (_Float16)X[(size_t)(Mbase + r) * DMODEL + k0 + c];
    }
    #pragma unroll
    for (int i = 0; i < 8; i++) {              // 64x32 B tile (W rows), f32 -> f16
      int idx = t + i * 256;
      int r = idx >> 5, c = idx & 31;
      Bs[r][c] = (_Float16)W[(size_t)(Nbase + r) * DMODEL + k0 + c];
    }
    __syncthreads();

    const int klo = hi * 8;    // A: lanes 0-15 K {0..7,16..23}; 16-31 K {8..15,24..31}
    const int kbB = hi * 8;    // B: lanes 0-15 K {0..7,16..23}; 16-31 K {8..15,24..31}
    v16h a[2], b[2];
    #pragma unroll
    for (int i = 0; i < 2; i++) {
      int m = wm * 32 + i * 16 + lo16;
      a[i] = make_frag(&As[m][klo], &As[m][klo + 16]);
    }
    #pragma unroll
    for (int j = 0; j < 2; j++) {
      int n = wn * 32 + j * 16 + lo16;
      b[j] = make_frag(&Bs[n][kbB], &Bs[n][kbB + 16]);
    }
    #pragma unroll
    for (int i = 0; i < 2; i++)
      #pragma unroll
      for (int j = 0; j < 2; j++)
        acc[i][j] = __builtin_amdgcn_wmma_f32_16x16x32_f16(
            false, a[i], false, b[j], (short)0, acc[i][j], false, false);
    __syncthreads();
  }

  // Epilogue: bias + store f16 into [B, H, S, DH] workspace layout.
  #pragma unroll
  for (int i = 0; i < 2; i++)
    #pragma unroll
    for (int j = 0; j < 2; j++) {
      int n = Nbase + wn * 32 + j * 16 + lo16;
      float bv_ = bias[n];
      int h = n >> 6, d = n & 63;
      #pragma unroll
      for (int r = 0; r < 8; r++) {
        int m = Mbase + wm * 32 + i * 16 + r + hi * 8;
        int bb = m >> 11, s = m & (SFULL - 1);
        Out[(((size_t)bb * NHEAD + h) * SFULL + s) * DHEAD + d] =
            (_Float16)(acc[i][j][r] + bv_);
      }
    }
}

// ---------------------------------------------------------------------------
// Flash attention with multiplicative post-softmax mask.
// Block = (qtile, b*H + h); 8 waves, each wave owns 16 query rows.
// K tile staged via CDNA5 async memory->LDS copies (ASYNCcnt path).
// ---------------------------------------------------------------------------
__global__ __launch_bounds__(256) void flash_attn_wmma(
    const _Float16* __restrict__ Qp, const _Float16* __restrict__ Kp,
    const _Float16* __restrict__ Vp, const float* __restrict__ mask,
    _Float16* __restrict__ Att)
{
  __shared__ __align__(16) _Float16 Ks[128][64];    // [key][d]
  __shared__ __align__(16) _Float16 Vs[64][128];    // [d][key]  (transposed)
  __shared__ __align__(16) _Float16 Ps[8][16][32];  // per-wave P staging

  const int t = threadIdx.x, lane = t & 31, wave = t >> 5;
  const int qtile = blockIdx.x;      // 16 tiles of 128 queries
  const int bh    = blockIdx.y;      // b*16 + h
  const int b = bh >> 4, h = bh & 15;
  const size_t bhoff = (size_t)bh * SFULL * DHEAD;
  const _Float16* Qb = Qp + bhoff;
  const _Float16* Kb = Kp + bhoff;
  const _Float16* Vb = Vp + bhoff;

  const int lo16 = lane & 15;
  const int hi   = (lane >> 4) & 1;
  const int qrow = qtile * 128 + wave * 16 + lo16;

  // Q fragments for DH=64 (two K=32 chunks), kept in registers.
  v16h qf[2];
  #pragma unroll
  for (int c = 0; c < 2; c++) {
    const _Float16* p = Qb + (size_t)qrow * DHEAD + c * 32 + hi * 8;
    qf[c] = make_frag(p, p + 16);
  }

  v8f o[4];
  #pragma unroll
  for (int nf = 0; nf < 4; nf++) { v8f z = {}; o[nf] = z; }
  float mr[8], lr[8];
  #pragma unroll
  for (int r = 0; r < 8; r++) { mr[r] = -3.0e38f; lr[r] = 0.0f; }

  for (int kb0 = 0; kb0 < SFULL; kb0 += 128) {
    // ---- stage K [128][64] via async memory->LDS b128 (no VGPR round-trip)
    #pragma unroll
    for (int i = 0; i < 4; i++) {
      int vi = t + i * 256;
      int key = vi >> 3, dc = (vi & 7) * 8;
      unsigned lds_addr = (unsigned)(size_t)&Ks[key][dc];
      unsigned goff = (unsigned)(((size_t)(kb0 + key) * DHEAD + dc) * sizeof(_Float16));
      async_ld_lds_b128(lds_addr, Kb, goff);
    }
    // ---- stage V transposed [d][key] (transpose needs the VGPR path)
    #pragma unroll
    for (int i = 0; i < 32; i++) {
      int idx = t + i * 256;
      int key = idx >> 6, d = idx & 63;
      Vs[d][key] = Vb[(size_t)(kb0 + key) * DHEAD + d];
    }
    // prefetch next chunk into L2 while this one is consumed
    if (kb0 + 128 < SFULL) {
      __builtin_prefetch(Kb + (size_t)(kb0 + 128 + (t >> 1)) * DHEAD, 0, 1);
      __builtin_prefetch(Vb + (size_t)(kb0 + 128 + (t >> 1)) * DHEAD, 0, 1);
    }
    wait_async_all();        // ASYNCcnt == 0 before publishing via barrier
    __syncthreads();

    #pragma unroll
    for (int j = 0; j < 4; j++) {              // 32 keys per step
      // ---- scores: two 16x16 tiles, 2 WMMA each (DH=64 = 2 chunks of 32)
      v8f s[2];
      #pragma unroll
      for (int tt = 0; tt < 2; tt++) {
        int keyn = j * 32 + tt * 16 + lo16;
        v8f accs = {};
        #pragma unroll
        for (int c = 0; c < 2; c++) {
          int off = c * 32 + hi * 8;
          v16h bf = make_frag(&Ks[keyn][off], &Ks[keyn][off + 16]);
          accs = __builtin_amdgcn_wmma_f32_16x16x32_f16(
              false, qf[c], false, bf, (short)0, accs, false, false);
        }
        s[tt] = accs * 0.125f;                 // 1/sqrt(DH)
      }

      // ---- online softmax update (16-lane butterflies), mask applied to P
      float alpha[8];
      #pragma unroll
      for (int r = 0; r < 8; r++) {
        float v0 = s[0][r], v1 = s[1][r];
        float mx = fmaxf(v0, v1);
        #pragma unroll
        for (int d = 1; d < 16; d <<= 1) mx = fmaxf(mx, __shfl_xor(mx, d, 32));
        float mnew = fmaxf(mr[r], mx);
        alpha[r] = __expf(mr[r] - mnew);
        mr[r] = mnew;
        float p0 = __expf(v0 - mnew);
        float p1 = __expf(v1 - mnew);
        float rs = p0 + p1;
        #pragma unroll
        for (int d = 1; d < 16; d <<= 1) rs += __shfl_xor(rs, d, 32);
        lr[r] = lr[r] * alpha[r] + rs;         // denominator is UNmasked
        int qg = qtile * 128 + wave * 16 + r + hi * 8;
        const float* mrow = mask + ((size_t)b * SFULL + qg) * SFULL + kb0 + j * 32;
        Ps[wave][r + hi * 8][lo16]      = (_Float16)(p0 * mrow[lo16]);
        Ps[wave][r + hi * 8][16 + lo16] = (_Float16)(p1 * mrow[16 + lo16]);
      }

      // ---- rescale O accumulators
      #pragma unroll
      for (int nf = 0; nf < 4; nf++)
        #pragma unroll
        for (int r = 0; r < 8; r++) o[nf][r] *= alpha[r];

      // ---- P (C-layout) -> A-layout via wave-private LDS (DS in-order per wave)
      v16h pf = make_frag(&Ps[wave][lo16][hi * 8], &Ps[wave][lo16][hi * 8 + 16]);

      // ---- O += P x V : 4 DH tiles of 16
      #pragma unroll
      for (int nf = 0; nf < 4; nf++) {
        int dh = nf * 16 + lo16;
        int koff = j * 32 + hi * 8;
        v16h vf = make_frag(&Vs[dh][koff], &Vs[dh][koff + 16]);
        o[nf] = __builtin_amdgcn_wmma_f32_16x16x32_f16(
            false, pf, false, vf, (short)0, o[nf], false, false);
      }
    }
    __syncthreads();
  }

  // Epilogue: atted [B, S, H*DH] f16
  #pragma unroll
  for (int nf = 0; nf < 4; nf++) {
    int col = h * DHEAD + nf * 16 + lo16;
    #pragma unroll
    for (int r = 0; r < 8; r++) {
      int sRow = qtile * 128 + wave * 16 + r + hi * 8;
      Att[((size_t)b * SFULL + sRow) * (NHEAD * DHEAD) + col] =
          (_Float16)(o[nf][r] / lr[r]);
    }
  }
}

// ---------------------------------------------------------------------------
// Output projection: out[m,n] (f32) = atted[m,:] (f16) @ Wm[n,:]^T + bm[n]
// ---------------------------------------------------------------------------
__global__ __launch_bounds__(256) void proj_out_wmma(
    const _Float16* __restrict__ X, const float* __restrict__ W,
    const float* __restrict__ bias, float* __restrict__ Out)
{
  __shared__ __align__(16) _Float16 As[128][32];
  __shared__ __align__(16) _Float16 Bs[64][32];

  const int t = threadIdx.x, lane = t & 31, wave = t >> 5;
  const int wm = wave & 3, wn = wave >> 2;
  const int Mbase = blockIdx.x * 128;
  const int Nbase = blockIdx.y * 64;
  const int lo16 = lane & 15;
  const int hi   = (lane >> 4) & 1;

  v8f acc[2][2];
  #pragma unroll
  for (int i = 0; i < 2; i++)
    #pragma unroll
    for (int j = 0; j < 2; j++) { v8f z = {}; acc[i][j] = z; }

  for (int k0 = 0; k0 < DMODEL; k0 += 32) {
    if (k0 + 32 < DMODEL) {
      __builtin_prefetch(&X[(size_t)(Mbase + (t >> 1)) * DMODEL + k0 + 32], 0, 1);
      if (t < 64)
        __builtin_prefetch(&W[(size_t)(Nbase + t) * DMODEL + k0 + 32], 0, 1);
    }
    #pragma unroll
    for (int i = 0; i < 16; i++) {
      int idx = t + i * 256;
      int r = idx >> 5, c = idx & 31;
      As[r][c] = X[(size_t)(Mbase + r) * DMODEL + k0 + c];
    }
    #pragma unroll
    for (int i = 0; i < 8; i++) {
      int idx = t + i * 256;
      int r = idx >> 5, c = idx & 31;
      Bs[r][c] = (_Float16)W[(size_t)(Nbase + r) * DMODEL + k0 + c];
    }
    __syncthreads();

    const int klo = hi * 8;
    const int kbB = hi * 8;
    v16h a[2], b[2];
    #pragma unroll
    for (int i = 0; i < 2; i++) {
      int m = wm * 32 + i * 16 + lo16;
      a[i] = make_frag(&As[m][klo], &As[m][klo + 16]);
    }
    #pragma unroll
    for (int j = 0; j < 2; j++) {
      int n = wn * 32 + j * 16 + lo16;
      b[j] = make_frag(&Bs[n][kbB], &Bs[n][kbB + 16]);
    }
    #pragma unroll
    for (int i = 0; i < 2; i++)
      #pragma unroll
      for (int j = 0; j < 2; j++)
        acc[i][j] = __builtin_amdgcn_wmma_f32_16x16x32_f16(
            false, a[i], false, b[j], (short)0, acc[i][j], false, false);
    __syncthreads();
  }

  #pragma unroll
  for (int i = 0; i < 2; i++)
    #pragma unroll
    for (int j = 0; j < 2; j++) {
      int n = Nbase + wn * 32 + j * 16 + lo16;
      float bv_ = bias[n];
      #pragma unroll
      for (int r = 0; r < 8; r++) {
        int m = Mbase + wm * 32 + i * 16 + r + hi * 8;
        Out[(size_t)m * DMODEL + n] = acc[i][j][r] + bv_;
      }
    }
}

// ---------------------------------------------------------------------------
extern "C" void kernel_launch(void* const* d_in, const int* in_sizes, int n_in,
                              void* d_out, int out_size, void* d_ws, size_t ws_size,
                              hipStream_t stream)
{
  const float* v    = (const float*)d_in[0];
  const float* k    = (const float*)d_in[1];
  const float* q    = (const float*)d_in[2];
  const float* mask = (const float*)d_in[3];
  const float* Wv   = (const float*)d_in[4];
  const float* bv   = (const float*)d_in[5];
  const float* Wk   = (const float*)d_in[6];
  const float* bk   = (const float*)d_in[7];
  const float* Wq   = (const float*)d_in[8];
  const float* bq   = (const float*)d_in[9];
  const float* Wm   = (const float*)d_in[10];
  const float* bm   = (const float*)d_in[11];
  float* out = (float*)d_out;

  // Workspace layout: Q | K | V (each [B,H,S,DH] f16) | atted [B,S,H*DH] f16
  const size_t elems = (size_t)2 * NHEAD * SFULL * DHEAD;   // 4M halves each
  _Float16* Qp = (_Float16*)d_ws;
  _Float16* Kp = Qp + elems;
  _Float16* Vp = Kp + elems;
  _Float16* At = Vp + elems;

  dim3 blk(256);
  dim3 gProj(32, 16);   // M=4096/128, N=1024/64
  proj_qkv_wmma<<<gProj, blk, 0, stream>>>(q, Wq, bq, Qp);
  proj_qkv_wmma<<<gProj, blk, 0, stream>>>(k, Wk, bk, Kp);
  proj_qkv_wmma<<<gProj, blk, 0, stream>>>(v, Wv, bv, Vp);

  dim3 gAtt(16, 32);    // 16 q-tiles x (B*H = 32)
  flash_attn_wmma<<<gAtt, blk, 0, stream>>>(Qp, Kp, Vp, mask, At);

  proj_out_wmma<<<gProj, blk, 0, stream>>>(At, Wm, bm, out);
}
